// LiquidConv50_50087908606184
// MI455X (gfx1250) — matrix operand order repaired and checked by reading, not compiled, not run on hardware
//
#include <hip/hip_runtime.h>
#include <hip/hip_bf16.h>
#include <math.h>

// ---------------------------------------------------------------------------
// LiquidConv (Conv1d + 2x LTC + FC) for MI455X / gfx1250.
//  Kernel 1: im2col conv as a real GEMM on the matrix pipe
//            (v_wmma_f32_16x16x32_f16, f32 accumulate) + bias + ReLU.
//  Kernel 2: persistent per-batch workgroup (512 thr = 16 wave32) runs the
//            full sequential scan with the two LTC layers software-pipelined
//            across time: waves 0-7 compute layer1 step t while waves 8-15
//            compute layer2 step t-1 (identical barrier skeletons, lockstep).
//            Recurrent edge params live in VGPRs (hot, per-unfold); sensory
//            params live in LDS (cold, per-step); recurrent state in LDS.
//            global_prefetch_b8 pre-touches the next activation row.
// Workspace: conv activations [B*T, 64] f32 = 4 MiB at d_ws offset 0.
// ---------------------------------------------------------------------------

typedef __attribute__((ext_vector_type(16))) _Float16 v16h;
typedef __attribute__((ext_vector_type(8)))  float    v8f;

#define BATCH   64
#define TLEN    256
#define CIN     8
#define COUT    50
#define KW      3
#define KIN     24      // CIN*KW im2col depth (padded to 32 for WMMA)
#define U1N     64
#define O1N     32
#define U2N     32
#define O2N     16
#define NUNF    6
#define EPSV    1e-8f
#define HSTRIDE 64      // padded conv-output row stride (floats)

__device__ __forceinline__ float sp_(float x) {
  return (x > 20.f) ? x : log1pf(__expf(x));      // softplus
}
__device__ __forceinline__ float sg_(float x) {
  return 1.f / (1.f + __expf(-x));                // sigmoid (v_exp_f32 + rcp)
}

// ---------------------------------------------------------------------------
// Kernel 1: conv via WMMA. One wave computes one 16(rows)x16(outs) tile.
// Rows = flattened (b,t); T=256 divisible by 16 so tiles never cross batch.
// A (16x32 f16): lane L -> m=L&15, half=L>>4; element e -> vgpr=e>>1,h=e&1;
//   K = (vgpr<4 ? 2*vgpr : 16+2*(vgpr-4)) + 8*half + h     (ISA 16-bit A map)
// B (32x16 f16): lane L -> n=L&15; K = (e<8 ? e : e+8) + 8*(L>>4)        (row striped)
// C (v8f): vgpr r -> m = r + 8*half, n = L&15.
// ---------------------------------------------------------------------------
__global__ __launch_bounds__(256)
void conv_wmma_kernel(const float* __restrict__ x,
                      const float* __restrict__ cw,
                      const float* __restrict__ cb,
                      float* __restrict__ h)
{
  const int lane    = threadIdx.x & 31;
  const int wid     = (blockIdx.x * blockDim.x + threadIdx.x) >> 5;
  const int rowTile = wid >> 2;          // 1024 row tiles over B*T
  const int colTile = wid & 3;           // 4 col tiles (64 >= 50 outs)
  const int half    = lane >> 4;
  const int ml      = lane & 15;

  // ---- A: im2col activations, causal left-pad 2 (branchless) ----
  const int row = rowTile * 16 + ml;
  const int bb  = row >> 8;
  const int tt  = row & 255;
  v16h a;
#pragma unroll
  for (int e = 0; e < 16; ++e) {
    const int vg = e >> 1, lo = e & 1;
    const int K  = (vg < 4 ? 2 * vg : 16 + 2 * (vg - 4)) + 8 * half + lo;
    const int Kc = (K < KIN) ? K : 0;
    const int ci = Kc / KW, dk = Kc % KW;
    const int ts  = tt - 2 + dk;
    const int tsc = (ts >= 0) ? ts : 0;
    const float xv = x[(bb * TLEN + tsc) * CIN + ci];   // always in-bounds
    a[e] = (_Float16)((K < KIN && ts >= 0) ? xv : 0.f);
  }

  // ---- B: weights, Wmat[k][o] = conv_w[o*24 + k] ----
  const int n  = lane & 15;
  const int o  = colTile * 16 + n;
  const int ks = half * 8;
  v16h bm;
#pragma unroll
  for (int e = 0; e < 16; ++e) {
    const int K = ks + (e < 8 ? e : e + 8);
    bm[e] = (_Float16)((K < KIN && o < COUT) ? cw[o * KIN + K] : 0.f);
  }

  v8f c = {};
  c = __builtin_amdgcn_wmma_f32_16x16x32_f16(false, a, false, bm,
                                             (short)0, c, false, false);

  const float bias = (o < COUT) ? cb[o] : 0.f;
#pragma unroll
  for (int r = 0; r < 8; ++r) {
    const int m    = r + 8 * half;
    const int orow = rowTile * 16 + m;
    float v = c[r] + bias;
    h[orow * HSTRIDE + o] = (v > 0.f) ? v : 0.f;   // cols >=50 are exact 0
  }
}

// ---------------------------------------------------------------------------
// Kernel 2: fused, time-pipelined two-layer LTC scan.
// grid = BATCH blocks x 512 threads (waves 0-7: layer1, waves 8-15: layer2).
// ---------------------------------------------------------------------------
struct LtcArgs {
  const float *gl1,*vl1,*cm1,*w1,*sg1,*mu1,*sw1,*ssg1,*smu1,*iw1,*ib1,*ow1,*ob1,*er1,*ser1;
  const float *gl2,*vl2,*cm2,*w2,*sg2,*mu2,*sw2,*ssg2,*smu2,*iw2,*ib2,*ow2,*ob2,*er2,*ser2;
  const float *fcw,*fcb,*mask1,*smask1,*mask2,*smask2;
  const float *h;        // conv output [B*T, HSTRIDE]
  float       *out;      // [B]
};

__global__ __launch_bounds__(512)
void ltc_fused_kernel(LtcArgs A)
{
  // recurrent state + reduction scratch
  __shared__ float v1[U1N], v2[U2N];
  __shared__ float wns1[U1N], wds1[U1N], wns2[U2N], wds2[U2N];
  __shared__ float pn1[256], pd1[256], pn2[256], pd2[256];
  __shared__ float xr[64];
  __shared__ float xr2buf[2][U2N];          // double-buffered layer1->layer2
  // per-neuron constants
  __shared__ float gl1s[U1N], gv1s[U1N], ct1s[U1N];
  __shared__ float gl2s[U2N], gv2s[U2N], ct2s[U2N];
  __shared__ float iw1s[64], ib1s[64];      // input affine (0-padded past 50)
  __shared__ float aouts[U2N], bouts[U2N];  // fused out1/in2 affine
  // layer1 sensory params (cold: once per step) -> LDS, conflict-free layout
  __shared__ float sSW[13 * 256], sSWE[13 * 256], sSS[13 * 256], sSM[13 * 256];

  const int tid = threadIdx.x;              // 0..511
  const bool g1 = tid < 256;                // layer-1 group
  const int b   = blockIdx.x;
  const int u1 = tid & 63,  blk1 = (tid >> 6) & 3;   // 4 blocks x 16 pre
  const int t2 = tid & 255;                 // layer-2 local id
  const int u2 = t2 & 31,   blk2 = t2 >> 5;          // 8 blocks x 4 pre

  // hot per-unfold params in VGPRs
  float W1[16], WE1[16], SG1[16], MU1[16];
  float W2[4], WE2[4], SG2[4], MU2[4], SW2[4], SWE2[4], SS2[4], SM2[4];

  if (g1) {
    // layer-1 recurrent edges: thread (u1, blk1) owns 16 pre-neurons
#pragma unroll
    for (int c = 0; c < 16; ++c) {
      const int idx = (blk1 * 16 + c) * U1N + u1;
      const float wp = sp_(A.w1[idx]) * A.mask1[idx];
      W1[c] = wp; WE1[c] = wp * A.er1[idx];
      SG1[c] = A.sg1[idx]; MU1[c] = A.mu1[idx];
    }
    // layer-1 sensory edges -> LDS (d = blk1 + 4*c, zero-padded past 50)
    for (int c = 0; c < 13; ++c) {
      const int d  = blk1 + 4 * c;
      const int dd = (d < COUT) ? d : 0;
      const int idx = dd * U1N + u1;
      const float live = (d < COUT) ? 1.f : 0.f;
      const float wp = sp_(A.sw1[idx]) * A.smask1[idx] * live;
      sSW [c * 256 + tid] = wp;
      sSWE[c * 256 + tid] = wp * A.ser1[idx];
      sSS [c * 256 + tid] = A.ssg1[idx] * live;
      sSM [c * 256 + tid] = A.smu1[idx] * live;
    }
    if (tid < U1N) {
      const float g = sp_(A.gl1[tid]);
      gl1s[tid] = g; gv1s[tid] = g * A.vl1[tid];
      ct1s[tid] = sp_(A.cm1[tid]) * (float)NUNF;
      v1[tid] = 0.f;
      iw1s[tid] = (tid < COUT) ? A.iw1[tid] : 0.f;
      ib1s[tid] = (tid < COUT) ? A.ib1[tid] : 0.f;
    }
    if (tid < U2N) {   // fused affine: xr2 = v1*aout + bout
      aouts[tid] = A.ow1[tid] * A.iw2[tid];
      bouts[tid] = A.ob1[tid] * A.iw2[tid] + A.ib2[tid];
    }
  } else {
    // layer-2 edges (sensory d-range == recurrent i-range for 32x32)
#pragma unroll
    for (int c = 0; c < 4; ++c) {
      const int idx = (blk2 * 4 + c) * U2N + u2;
      const float wp = sp_(A.w2[idx]) * A.mask2[idx];
      W2[c] = wp; WE2[c] = wp * A.er2[idx];
      SG2[c] = A.sg2[idx]; MU2[c] = A.mu2[idx];
      const float sw = sp_(A.sw2[idx]) * A.smask2[idx];
      SW2[c] = sw; SWE2[c] = sw * A.ser2[idx];
      SS2[c] = A.ssg2[idx]; SM2[c] = A.smu2[idx];
    }
    if (t2 < U2N) {
      const float g = sp_(A.gl2[t2]);
      gl2s[t2] = g; gv2s[t2] = g * A.vl2[t2];
      ct2s[t2] = sp_(A.cm2[t2]) * (float)NUNF;
      v2[t2] = 0.f;
    }
  }
  __syncthreads();

  // Pipelined scan: iteration t -> layer1 computes step t (t < T),
  //                                layer2 computes step t-1 (t >= 1).
  for (int t = 0; t <= TLEN; ++t) {
    // ---- stage conv row (+input affine) for layer1; prefetch next row ----
    if (g1 && t < TLEN && tid < 64) {
      const float xv = A.h[(b * TLEN + t) * HSTRIDE + tid];
      xr[tid] = xv * iw1s[tid] + ib1s[tid];
    }
    if (tid == 0 && t + 1 < TLEN)
      __builtin_prefetch(&A.h[(b * TLEN + t + 1) * HSTRIDE], 0, 1);
    __syncthreads();

    // ---- sensory partials ----
    if (g1) {
      if (t < TLEN) {
        float an = 0.f, ad = 0.f;
#pragma unroll
        for (int c = 0; c < 13; ++c) {
          const int d = blk1 + 4 * c;              // <=51, xr affine zeroed
          const float s = sg_(sSS[c*256+tid] * (xr[d] - sSM[c*256+tid]));
          an += sSWE[c*256+tid] * s; ad += sSW[c*256+tid] * s;
        }
        pn1[tid] = an; pd1[tid] = ad;
      }
    } else if (t >= 1) {
      const float* xin = xr2buf[(t - 1) & 1];
      float an = 0.f, ad = 0.f;
#pragma unroll
      for (int c = 0; c < 4; ++c) {
        const float s = sg_(SS2[c] * (xin[blk2 * 4 + c] - SM2[c]));
        an += SWE2[c] * s; ad += SW2[c] * s;
      }
      pn2[t2] = an; pd2[t2] = ad;
    }
    __syncthreads();

    // ---- sensory reductions ----
    if (g1) {
      if (t < TLEN && tid < U1N) {
        wns1[tid] = pn1[tid] + pn1[tid+64] + pn1[tid+128] + pn1[tid+192];
        wds1[tid] = pd1[tid] + pd1[tid+64] + pd1[tid+128] + pd1[tid+192];
      }
    } else if (t >= 1 && t2 < U2N) {
      float an = 0.f, ad = 0.f;
#pragma unroll
      for (int k = 0; k < 8; ++k) { an += pn2[t2 + 32*k]; ad += pd2[t2 + 32*k]; }
      wns2[t2] = an; wds2[t2] = ad;
    }
    __syncthreads();

    // ---- 6 fused-solver unfolds (both layers in lockstep) ----
    for (int nuf = 0; nuf < NUNF; ++nuf) {
      if (g1) {
        if (t < TLEN) {
          float an = 0.f, ad = 0.f;
#pragma unroll
          for (int c = 0; c < 16; ++c) {
            const float s = sg_(SG1[c] * (v1[blk1 * 16 + c] - MU1[c]));
            an += WE1[c] * s; ad += W1[c] * s;
          }
          pn1[tid] = an; pd1[tid] = ad;
        }
      } else if (t >= 1) {
        float an = 0.f, ad = 0.f;
#pragma unroll
        for (int c = 0; c < 4; ++c) {
          const float s = sg_(SG2[c] * (v2[blk2 * 4 + c] - MU2[c]));
          an += WE2[c] * s; ad += W2[c] * s;
        }
        pn2[t2] = an; pd2[t2] = ad;
      }
      __syncthreads();
      if (g1) {
        if (t < TLEN && tid < U1N) {
          const float num = pn1[tid]+pn1[tid+64]+pn1[tid+128]+pn1[tid+192] + wns1[tid];
          const float den = pd1[tid]+pd1[tid+64]+pd1[tid+128]+pd1[tid+192] + wds1[tid];
          v1[tid] = (ct1s[tid] * v1[tid] + gv1s[tid] + num)
                  / (ct1s[tid] + gl1s[tid] + den + EPSV);
        }
      } else if (t >= 1 && t2 < U2N) {
        float num = wns2[t2], den = wds2[t2];
#pragma unroll
        for (int k = 0; k < 8; ++k) { num += pn2[t2 + 32*k]; den += pd2[t2 + 32*k]; }
        v2[t2] = (ct2s[t2] * v2[t2] + gv2s[t2] + num)
               / (ct2s[t2] + gl2s[t2] + den + EPSV);
      }
      __syncthreads();
    }

    // ---- publish layer1 motor output for layer2's next iteration ----
    // (next consumer read is after the loop-top barrier of iteration t+1;
    //  next overwrite of this buffer is at iteration t+2 -> race-free)
    if (g1 && t < TLEN && tid < U2N)
      xr2buf[t & 1][tid] = v1[tid] * aouts[tid] + bouts[tid];
  }

  __syncthreads();
  // ---- final: motor outputs -> FC -> sigmoid ----
  if (tid == 256) {
    float acc = A.fcb[0];
#pragma unroll
    for (int j = 0; j < O2N; ++j)
      acc += A.fcw[j] * (v2[j] * A.ow2[j] + A.ob2[j]);
    A.out[b] = sg_(acc);
  }
}

// ---------------------------------------------------------------------------
extern "C" void kernel_launch(void* const* d_in, const int* in_sizes, int n_in,
                              void* d_out, int out_size, void* d_ws, size_t ws_size,
                              hipStream_t stream)
{
  (void)in_sizes; (void)n_in; (void)out_size; (void)ws_size;
  auto in = [&](int i) { return (const float*)d_in[i]; };

  // workspace: conv activations [B*T, HSTRIDE] f32 (4 MiB)
  float* h = (float*)d_ws;

  // Kernel 1: 1024 row tiles x 4 col tiles = 4096 waves, 8 waves/block
  conv_wmma_kernel<<<512, 256, 0, stream>>>(in(0), in(1), in(2), h);

  LtcArgs a;
  // p1 (indices 3..17): gleak vleak cm w sigma mu sw ssigma smu in_w in_b out_w out_b erev serev
  a.gl1 = in(3);  a.vl1 = in(4);  a.cm1 = in(5);  a.w1  = in(6);  a.sg1 = in(7);
  a.mu1 = in(8);  a.sw1 = in(9);  a.ssg1= in(10); a.smu1= in(11); a.iw1 = in(12);
  a.ib1 = in(13); a.ow1 = in(14); a.ob1 = in(15); a.er1 = in(16); a.ser1= in(17);
  // p2 (indices 18..32)
  a.gl2 = in(18); a.vl2 = in(19); a.cm2 = in(20); a.w2  = in(21); a.sg2 = in(22);
  a.mu2 = in(23); a.sw2 = in(24); a.ssg2= in(25); a.smu2= in(26); a.iw2 = in(27);
  a.ib2 = in(28); a.ow2 = in(29); a.ob2 = in(30); a.er2 = in(31); a.ser2= in(32);
  a.fcw = in(33); a.fcb = in(34);
  a.mask1 = in(35); a.smask1 = in(36); a.mask2 = in(37); a.smask2 = in(38);
  a.h = h; a.out = (float*)d_out;

  ltc_fused_kernel<<<BATCH, 512, 0, stream>>>(a);
}
